// SceneSayerSDE_32950989095108
// MI455X (gfx1250) — compile-verified, matrix operand order checked by reading, not run on hardware
//
#include <hip/hip_runtime.h>
#include <hip/hip_bf16.h>
#include <math.h>

// ---------------------------------------------------------------------------
// Types for CDNA5 WMMA (wave32, 16x16x32 bf16 -> f32)
// ---------------------------------------------------------------------------
typedef __bf16 bf16;
typedef __attribute__((ext_vector_type(8)))  __bf16 bf16x8;
typedef __attribute__((ext_vector_type(16))) __bf16 bf16x16;
typedef __attribute__((ext_vector_type(8)))  float  f32x8;

__device__ __forceinline__ bf16 f2bf(float f) {
  unsigned u = __builtin_bit_cast(unsigned, f);
  unsigned r = u + 0x7FFFu + ((u >> 16) & 1u);   // round-to-nearest-even
  unsigned short h = (unsigned short)(r >> 16);
  return __builtin_bit_cast(bf16, h);
}

// CDNA5 async global->LDS copy (16B per lane), tracked by ASYNCcnt.
__device__ __forceinline__ void async_load_b128(unsigned lds_off,
                                                unsigned long long gaddr) {
  asm volatile("global_load_async_to_lds_b128 %0, %1, off"
               :: "v"(lds_off), "v"(gaddr) : "memory");
}
__device__ __forceinline__ void wait_asynccnt0() {
  asm volatile("s_wait_asynccnt 0x0" ::: "memory");
}

// Problem constants
#define NF   64
#define NP   16
#define NDIM 1936
#define NH   8
#define NHD  242
#define NFF  2048
#define NROW 1024        /* F*P */

// ---------------------------------------------------------------------------
// Generic bf16 WMMA GEMM:  C[M,N] = act(A[M,K] @ B[K,N] + bias)
// 256 threads (8 waves), block tile 128x128, K-step 32, double-buffered LDS.
// Wave grid 4(M) x 2(N): each wave computes 32x64 via 2x4 WMMA fragments.
// A tile staged with GLOBAL_LOAD_ASYNC_TO_LDS_B128; B tile staged transposed.
// Optionally writes a bf16 mirror of C for chaining into the next GEMM.
// ---------------------------------------------------------------------------
#define TM 128
#define TN 128
#define TK 32
#define LDK 40           /* padded LDS row stride (elems); 80B keeps 16B align */

__global__ __launch_bounds__(256)
void gemm_bf16_kernel(const bf16* __restrict__ A, const bf16* __restrict__ B,
                      const float* __restrict__ bias, float* __restrict__ C,
                      bf16* __restrict__ Cbf, int M, int N, int K, int act)
{
  __shared__ bf16 As[2][TM][LDK];   // row-major A tiles
  __shared__ bf16 Bs[2][TN][LDK];   // transposed B tiles: Bs[buf][n][k]

  const int tid  = threadIdx.x;
  const int wave = tid >> 5;
  const int lane = tid & 31;
  const int wm   = wave >> 1;       // 0..3  -> 32-row slice
  const int wn   = wave & 1;        // 0..1  -> 64-col slice
  const int m0 = blockIdx.y * TM;
  const int n0 = blockIdx.x * TN;

  const bf16 bz = __builtin_bit_cast(bf16, (unsigned short)0);

  f32x8 acc[2][4];
#pragma unroll
  for (int u = 0; u < 2; ++u)
#pragma unroll
    for (int t = 0; t < 4; ++t)
#pragma unroll
      for (int i = 0; i < 8; ++i) acc[u][t][i] = 0.0f;

  // ---- tile staging ----
  auto stage = [&](int buf, int k0) {
    // A: 128 rows x 32 k = 512 16B-chunks; async copy straight into LDS
    for (int c = tid; c < (TM * TK / 8); c += 256) {
      int row = c >> 2;
      int ko  = (c & 3) * 8;
      int gm = m0 + row, gk = k0 + ko;
      unsigned lo = (unsigned)(uintptr_t)&As[buf][row][ko];
      if (gm < M && (gk + 8) <= K) {
        async_load_b128(lo, (unsigned long long)(uintptr_t)(A + (size_t)gm * K + gk));
      } else {
        bf16x8 z;
#pragma unroll
        for (int j = 0; j < 8; ++j) z[j] = bz;
        *(bf16x8*)&As[buf][row][ko] = z;
      }
    }
    // B: 32 k-rows x 128 n = 512 chunks, scattered transposed into LDS
    for (int c = tid; c < (TK * TN / 8); c += 256) {
      int krow = c >> 4;           // 16 chunks per k-row
      int no   = (c & 15) * 8;
      bf16x8 v;
#pragma unroll
      for (int j = 0; j < 8; ++j) v[j] = bz;
      int gk = k0 + krow, gn = n0 + no;
      if (gk < K && (gn + 8) <= N)
        v = *(const bf16x8*)(B + (size_t)gk * N + gn);
#pragma unroll
      for (int j = 0; j < 8; ++j) Bs[buf][no + j][krow] = v[j];
    }
  };

  // ---- WMMA compute on one buffer ----
  auto compute = [&](int buf) {
    const int rsel = lane & 15;
    const int kh   = lane >> 4;
    bf16x16 a[2];
#pragma unroll
    for (int u = 0; u < 2; ++u) {
      bf16x8 alo = *(const bf16x8*)&As[buf][wm * 32 + u * 16 + rsel][8 * kh];
      bf16x8 ahi = *(const bf16x8*)&As[buf][wm * 32 + u * 16 + rsel][16 + 8 * kh];
#pragma unroll
      for (int i = 0; i < 8; ++i) { a[u][i] = alo[i]; a[u][8 + i] = ahi[i]; }
    }
#pragma unroll
    for (int t = 0; t < 4; ++t) {
      bf16x8 blo = *(const bf16x8*)&Bs[buf][wn * 64 + t * 16 + rsel][8 * kh];
      bf16x8 bhi = *(const bf16x8*)&Bs[buf][wn * 64 + t * 16 + rsel][16 + 8 * kh];
      bf16x16 b;
#pragma unroll
      for (int i = 0; i < 8; ++i) { b[i] = blo[i]; b[8 + i] = bhi[i]; }
#pragma unroll
      for (int u = 0; u < 2; ++u)
        acc[u][t] = __builtin_amdgcn_wmma_f32_16x16x32_bf16(
                      false, a[u], false, b, (short)0, acc[u][t], false, false);
    }
  };

  const int kt = (K + TK - 1) / TK;
  stage(0, 0);
  wait_asynccnt0();
  __syncthreads();
  int buf = 0;
  for (int kb = 0; kb < kt; ++kb) {
    if (kb + 1 < kt) stage(buf ^ 1, (kb + 1) * TK);   // overlap with compute
    compute(buf);
    wait_asynccnt0();
    __syncthreads();
    buf ^= 1;
  }

  // ---- epilogue: bias + act + store (f32, optional bf16 mirror) ----
  const int colL = lane & 15;
  const int mh   = lane >> 4;
#pragma unroll
  for (int u = 0; u < 2; ++u)
#pragma unroll
    for (int t = 0; t < 4; ++t) {
      int n = n0 + wn * 64 + t * 16 + colL;
      if (n >= N) continue;
      float bv = bias ? bias[n] : 0.0f;
#pragma unroll
      for (int r = 0; r < 8; ++r) {
        int m = m0 + wm * 32 + u * 16 + mh * 8 + r;
        if (m >= M) continue;
        float v = acc[u][t][r] + bv;
        if (act == 1) v = fmaxf(v, 0.0f);
        C[(size_t)m * N + n] = v;
        if (Cbf) Cbf[(size_t)m * N + n] = f2bf(v);
      }
    }
}

// ---------------------------------------------------------------------------
// Elementwise / pack / reduction kernels
// ---------------------------------------------------------------------------
__global__ void convert_f2b_kernel(const float* __restrict__ src,
                                   bf16* __restrict__ dst, int n) {
  int i = blockIdx.x * blockDim.x + threadIdx.x;
  if (i < n) dst[i] = f2bf(src[i]);
}

// src (R,C) f32 row-major -> dst (C,R) bf16
__global__ void transpose_cvt_kernel(const float* __restrict__ src,
                                     bf16* __restrict__ dst, int R, int C) {
  int i = blockIdx.x * blockDim.x + threadIdx.x;
  int total = R * C;
  if (i >= total) return;
  int r = i % R, c = i / R;
  dst[i] = f2bf(src[(size_t)r * C + c]);
}

// gather feature rows by pair_idx column into bf16 matrix (1024 x 2376)
__global__ void gather_bf16_kernel(const float* __restrict__ feat,
                                   const int* __restrict__ pair, int colsel,
                                   bf16* __restrict__ dst) {
  int r = blockIdx.x;
  int src = pair[2 * r + colsel];
  for (int d = threadIdx.x; d < 2376; d += 256)
    dst[(size_t)r * 2376 + d] = f2bf(feat[(size_t)src * 2376 + d]);
}

// union_feat (1024,1024,49) NCHW -> A matrix [(n*49+hw), c] bf16
__global__ void union_pack_kernel(const float* __restrict__ uf,
                                  bf16* __restrict__ dst) {
  int i = blockIdx.x * blockDim.x + threadIdx.x;
  const int total = 1024 * 1024 * 49;
  if (i >= total) return;
  int c = i & 1023;
  int row = i >> 10;
  int hw = row % 49, n = row / 49;
  dst[(size_t)i] = f2bf(uf[((size_t)n * 1024 + c) * 49 + hw]);
}

// fused conv1(7x7 s2 p3) + bias + relu + bn1 + maxpool(3x3 s2 p1)
// in: (1024,2,27,27)  out: (1024,128,7,7)
__global__ void conv1_pool_kernel(const float* __restrict__ in,
                                  const float* __restrict__ w,
                                  const float* __restrict__ bias,
                                  const float* __restrict__ g,
                                  const float* __restrict__ bct,
                                  float* __restrict__ out) {
  int idx = blockIdx.x * blockDim.x + threadIdx.x;
  const int total = 1024 * 128 * 49;
  if (idx >= total) return;
  int pw = idx % 7, ph = (idx / 7) % 7, c = (idx / 49) % 128, n = idx / (49 * 128);
  float scale = g[c] * rsqrtf(1.0f + 1e-5f);
  float mx = -1e30f;
  for (int wy = 0; wy < 3; ++wy)
    for (int wx = 0; wx < 3; ++wx) {
      int oy = ph * 2 + wy - 1, ox = pw * 2 + wx - 1;
      if (oy < 0 || oy >= 14 || ox < 0 || ox >= 14) continue;
      float acc = bias[c];
      for (int ic = 0; ic < 2; ++ic)
        for (int ky = 0; ky < 7; ++ky) {
          int iy = oy * 2 + ky - 3; if (iy < 0 || iy >= 27) continue;
          for (int kx = 0; kx < 7; ++kx) {
            int ix = ox * 2 + kx - 3; if (ix < 0 || ix >= 27) continue;
            acc += in[(((size_t)n * 2 + ic) * 27 + iy) * 27 + ix] *
                   w[(((size_t)c * 2 + ic) * 7 + ky) * 7 + kx];
          }
        }
      float bnv = fmaxf(acc, 0.0f) * scale + bct[c];
      mx = fmaxf(mx, bnv);
    }
  out[idx] = mx;
}

// pooled (1024,128,7,7) -> im2col rows (50176, 1152) bf16 (3x3 pad 1)
__global__ void im2col_kernel(const float* __restrict__ pooled,
                              bf16* __restrict__ dst) {
  long long i = (long long)blockIdx.x * blockDim.x + threadIdx.x;
  const long long total = 50176LL * 1152LL;
  if (i >= total) return;
  int col = (int)(i % 1152);
  long long row = i / 1152;
  int kw = col % 3, kh = (col / 3) % 3, c = col / 9;
  int ow = (int)(row % 7), oh = (int)((row / 7) % 7), n = (int)(row / 49);
  int iy = oh + kh - 1, ix = ow + kw - 1;
  float v = (iy < 0 || iy >= 7 || ix < 0 || ix >= 7)
              ? 0.0f
              : pooled[(((size_t)n * 128 + c) * 7 + iy) * 7 + ix];
  dst[i] = f2bf(v);
}

// vr_in[n][c*49+hw] = bf16( u_mat[(n*49+hw),c] + bn2(relu(conv2out[(n*49+hw),c])) )
__global__ void combine_vr_kernel(const float* __restrict__ u_mat,
                                  const float* __restrict__ c2,
                                  const float* __restrict__ g,
                                  const float* __restrict__ b,
                                  bf16* __restrict__ vr_in) {
  int i = blockIdx.x * blockDim.x + threadIdx.x;
  const int total = 1024 * 12544;
  if (i >= total) return;
  int col = i % 12544, n = i / 12544;
  int c = col / 49, hw = col % 49;
  size_t src = ((size_t)n * 49 + hw) * 256 + c;
  float s = fmaxf(c2[src], 0.0f) * (g[c] * rsqrtf(1.0f + 1e-5f)) + b[c];
  vr_in[(size_t)i] = f2bf(u_mat[src] + s);
}

// rel = concat(subj, obj, vr, emb1[lbl_s], emb2[lbl_o])  -> f32 + bf16
__global__ void build_rel_kernel(const float* __restrict__ subj,
                                 const float* __restrict__ obj,
                                 const float* __restrict__ vr,
                                 const float* __restrict__ emb1,
                                 const float* __restrict__ emb2,
                                 const int* __restrict__ pair,
                                 const int* __restrict__ pred,
                                 float* __restrict__ X, bf16* __restrict__ Xb) {
  int r = blockIdx.x;
  int si = pair[2 * r], oi = pair[2 * r + 1];
  int l1 = pred[si], l2 = pred[oi];
  size_t base = (size_t)r * NDIM;
  for (int d = threadIdx.x; d < NDIM; d += 256) {
    float v;
    if      (d < 512)  v = subj[(size_t)r * 512 + d];
    else if (d < 1024) v = obj[(size_t)r * 512 + (d - 512)];
    else if (d < 1536) v = vr[(size_t)r * 512 + (d - 1024)];
    else if (d < 1736) v = emb1[(size_t)l1 * 200 + (d - 1536)];
    else               v = emb2[(size_t)l2 * 200 + (d - 1736)];
    X[base + d] = v;
    Xb[base + d] = f2bf(v);
  }
}

// softmax attention: Q,K,V laid out (B*S, D) with head h at cols [h*Hd, (h+1)*Hd)
__global__ void attn_kernel(const float* __restrict__ Q,
                            const float* __restrict__ K,
                            const float* __restrict__ V,
                            float* __restrict__ O, int Bb, int S, int causal) {
  __shared__ float sc[64];
  int bid = blockIdx.x;
  int qi = bid % S;
  int h  = (bid / S) % NH;
  int b  = bid / (S * NH);
  int tid = threadIdx.x, wave = tid >> 5, lane = tid & 31;
  const float scale = rsqrtf((float)NHD);
  const float* qrow = Q + ((size_t)(b * S + qi)) * NDIM + h * NHD;

  for (int ki = wave; ki < S; ki += 8) {
    const float* krow = K + ((size_t)(b * S + ki)) * NDIM + h * NHD;
    float p = 0.0f;
    for (int d = lane; d < NHD; d += 32) p += qrow[d] * krow[d];
    for (int o = 16; o > 0; o >>= 1) p += __shfl_xor(p, o, 32);
    if (lane == 0) {
      float v = p * scale;
      if (causal && ki > qi) v = -1e9f;
      sc[ki] = v;
    }
  }
  __syncthreads();
  if (wave == 0) {
    float m = -1e30f;
    for (int i = lane; i < S; i += 32) m = fmaxf(m, sc[i]);
    for (int o = 16; o > 0; o >>= 1) m = fmaxf(m, __shfl_xor(m, o, 32));
    float s = 0.0f;
    for (int i = lane; i < S; i += 32) { float e = __expf(sc[i] - m); sc[i] = e; s += e; }
    for (int o = 16; o > 0; o >>= 1) s += __shfl_xor(s, o, 32);
    float inv = 1.0f / s;
    for (int i = lane; i < S; i += 32) sc[i] *= inv;
  }
  __syncthreads();
  float* orow = O + ((size_t)(b * S + qi)) * NDIM + h * NHD;
  for (int d = tid; d < NHD; d += 256) {
    float acc = 0.0f;
    for (int ki = 0; ki < S; ++ki)
      acc += sc[ki] * V[((size_t)(b * S + ki)) * NDIM + h * NHD + d];
    orow[d] = acc;
  }
}

// Y = LN(X + R) * g + b   (R optional); writes f32 and optional bf16 mirror
__global__ void ln_res_kernel(const float* __restrict__ X,
                              const float* __restrict__ Rr,
                              const float* __restrict__ g,
                              const float* __restrict__ bb,
                              float* __restrict__ Y, bf16* __restrict__ Yb,
                              int D) {
  int row = blockIdx.x;
  size_t base = (size_t)row * D;
  __shared__ float red[8];
  __shared__ float stat[2];
  int tid = threadIdx.x, wave = tid >> 5, lane = tid & 31;

  float s = 0.0f;
  for (int d = tid; d < D; d += 256) {
    float v = X[base + d]; if (Rr) v += Rr[base + d];
    s += v;
  }
  for (int o = 16; o > 0; o >>= 1) s += __shfl_xor(s, o, 32);
  if (lane == 0) red[wave] = s;
  __syncthreads();
  if (tid == 0) { float t = 0; for (int i = 0; i < 8; ++i) t += red[i]; stat[0] = t / D; }
  __syncthreads();
  float mean = stat[0];

  float vs = 0.0f;
  for (int d = tid; d < D; d += 256) {
    float v = X[base + d]; if (Rr) v += Rr[base + d];
    v -= mean; vs += v * v;
  }
  for (int o = 16; o > 0; o >>= 1) vs += __shfl_xor(vs, o, 32);
  __syncthreads();
  if (lane == 0) red[wave] = vs;
  __syncthreads();
  if (tid == 0) { float t = 0; for (int i = 0; i < 8; ++i) t += red[i]; stat[1] = rsqrtf(t / D + 1e-5f); }
  __syncthreads();
  float inv = stat[1];

  for (int d = tid; d < D; d += 256) {
    float v = X[base + d]; if (Rr) v += Rr[base + d];
    float y = (v - mean) * inv * g[d] + bb[d];
    Y[base + d] = y;
    if (Yb) Yb[base + d] = f2bf(y);
  }
}

// (F,P,D)->(P,F,D) transpose + sinusoidal pos-encoding over f; f32 + bf16
__global__ void pos_transpose_kernel(const float* __restrict__ Xl,
                                     float* __restrict__ Xg,
                                     bf16* __restrict__ Xgb) {
  int row = blockIdx.x;          // output row = p*64 + f
  int f = row & 63, p = row >> 6;
  size_t src = ((size_t)(f * NP + p)) * NDIM;
  size_t dst = (size_t)row * NDIM;
  const float c = -logf(10000.0f) / (float)NDIM;
  for (int d = threadIdx.x; d < NDIM; d += 256) {
    int i = d >> 1;
    float ang = (float)f * __expf(c * (2.0f * (float)i));
    float pe = (d & 1) ? __cosf(ang) : __sinf(ang);
    float v = Xl[src + d] + pe;
    Xg[dst + d] = v;
    Xgb[dst + d] = f2bf(v);
  }
}

// un-transpose (P,F,D)->(F,P,D) into global_output region of d_out
__global__ void write_go_kernel(const float* __restrict__ Xg,
                                float* __restrict__ go) {
  int row = blockIdx.x;          // n = f*16 + p
  int f = row >> 4, p = row & 15;
  size_t src = ((size_t)(p * NF + f)) * NDIM;
  size_t dst = (size_t)row * NDIM;
  for (int d = threadIdx.x; d < NDIM; d += 256) go[dst + d] = Xg[src + d];
}

// heads: att (3) | sigmoid spat (6) | sigmoid cont (17)
__global__ void heads_kernel(const float* __restrict__ go,
                             const float* __restrict__ aw, const float* __restrict__ ab,
                             const float* __restrict__ sw, const float* __restrict__ sb,
                             const float* __restrict__ cw, const float* __restrict__ cb,
                             float* __restrict__ out) {
  int j = blockIdx.x, row = blockIdx.y;
  const float* wb; const float* bp; int stride, col, sig;
  if (j < 3)      { wb = aw; bp = ab; stride = 3;  col = j;      sig = 0; }
  else if (j < 9) { wb = sw; bp = sb; stride = 6;  col = j - 3;  sig = 1; }
  else            { wb = cw; bp = cb; stride = 17; col = j - 9;  sig = 1; }
  float s = 0.0f;
  for (int d = threadIdx.x; d < NDIM; d += 64)
    s += go[(size_t)row * NDIM + d] * wb[(size_t)d * stride + col];
  for (int o = 16; o > 0; o >>= 1) s += __shfl_xor(s, o, 32);
  __shared__ float red[2];
  if ((threadIdx.x & 31) == 0) red[threadIdx.x >> 5] = s;
  __syncthreads();
  if (threadIdx.x == 0) {
    float v = red[0] + red[1] + bp[col];
    if (sig) v = 1.0f / (1.0f + __expf(-v));
    if (j < 3)      out[row * 3 + j] = v;
    else if (j < 9) out[3072 + row * 6 + (j - 3)] = v;
    else            out[9216 + row * 17 + (j - 9)] = v;
  }
}

// ---------------------------------------------------------------------------
// Host orchestration
// ---------------------------------------------------------------------------
extern "C" void kernel_launch(void* const* d_in, const int* in_sizes, int n_in,
                              void* d_out, int out_size, void* d_ws, size_t ws_size,
                              hipStream_t stream) {
  (void)in_sizes; (void)n_in; (void)out_size; (void)ws_size;
  auto FP = [&](int i) { return (const float*)d_in[i]; };
  const float* features   = FP(0);
  const float* union_feat = FP(1);
  const float* spatial    = FP(2);
  const int*   pair_idx   = (const int*)d_in[91];
  const int*   pred_lab   = (const int*)d_in[92];

  // --- workspace allocator ---
  char* ws = (char*)d_ws;
  size_t off = 0;
  auto alloc = [&](size_t bytes) -> void* {
    size_t a = (off + 255) & ~(size_t)255;
    off = a + bytes;
    return (void*)(ws + a);
  };

  const int D = NDIM;
  // bf16 weights
  bf16 *Wq[4], *Wk[4], *Wv[4], *Wo[4], *W1[4], *W2[4];
  for (int l = 0; l < 4; ++l) {
    Wq[l] = (bf16*)alloc((size_t)D * D * 2);
    Wk[l] = (bf16*)alloc((size_t)D * D * 2);
    Wv[l] = (bf16*)alloc((size_t)D * D * 2);
    Wo[l] = (bf16*)alloc((size_t)D * D * 2);
    W1[l] = (bf16*)alloc((size_t)D * NFF * 2);
    W2[l] = (bf16*)alloc((size_t)NFF * D * 2);
  }
  bf16* subjW = (bf16*)alloc((size_t)2376 * 512 * 2);
  bf16* objW  = (bf16*)alloc((size_t)2376 * 512 * 2);
  bf16* vrW   = (bf16*)alloc((size_t)12544 * 512 * 2);
  bf16* uW    = (bf16*)alloc((size_t)1024 * 256 * 2);
  bf16* c2W   = (bf16*)alloc((size_t)1152 * 256 * 2);
  // activations
  bf16*  subjA  = (bf16*)alloc((size_t)1024 * 2376 * 2);
  bf16*  objA   = (bf16*)alloc((size_t)1024 * 2376 * 2);
  float* subjO  = (float*)alloc((size_t)1024 * 512 * 4);
  float* objO   = (float*)alloc((size_t)1024 * 512 * 4);
  float* vrO    = (float*)alloc((size_t)1024 * 512 * 4);
  bf16*  uA     = (bf16*)alloc((size_t)50176 * 1024 * 2);  // reused as conv2-out (f32, 51MB<102MB)
  float* u_mat  = (float*)alloc((size_t)50176 * 256 * 4);
  float* pooled = (float*)alloc((size_t)1024 * 128 * 49 * 4);
  bf16*  colB   = (bf16*)alloc((size_t)50176 * 1152 * 2);
  float* c2out  = (float*)uA;                              // alias (uA dead by then)
  bf16*  vrIn   = (bf16*)alloc((size_t)1024 * 12544 * 2);
  // transformer buffers
  float* X1  = (float*)alloc((size_t)NROW * D * 4);
  bf16*  X1b = (bf16*) alloc((size_t)NROW * D * 2);
  float* X2  = (float*)alloc((size_t)NROW * D * 4);
  bf16*  X2b = (bf16*) alloc((size_t)NROW * D * 2);
  float* Qb  = (float*)alloc((size_t)NROW * D * 4);
  float* Kb  = (float*)alloc((size_t)NROW * D * 4);
  float* Vb  = (float*)alloc((size_t)NROW * D * 4);
  float* Ob  = (float*)alloc((size_t)NROW * D * 4);
  bf16*  Obf = (bf16*) alloc((size_t)NROW * D * 2);
  float* Pb  = (float*)alloc((size_t)NROW * D * 4);
  float* Yb  = (float*)alloc((size_t)NROW * D * 4);
  bf16*  Ybf = (bf16*) alloc((size_t)NROW * D * 2);
  float* Hb  = (float*)alloc((size_t)NROW * NFF * 4);
  bf16*  Hbf = (bf16*) alloc((size_t)NROW * NFF * 2);
  float* Fb  = (float*)alloc((size_t)NROW * D * 4);

  auto cvt = [&](const void* src, bf16* dst, int n) {
    convert_f2b_kernel<<<(n + 255) / 256, 256, 0, stream>>>((const float*)src, dst, n);
  };
  auto gemm = [&](const bf16* A, const bf16* B, const float* bias, float* C,
                  bf16* Cb, int M, int N, int K, int act) {
    dim3 grid((N + TN - 1) / TN, (M + TM - 1) / TM);
    gemm_bf16_kernel<<<grid, 256, 0, stream>>>(A, B, bias, C, Cb, M, N, K, act);
  };

  // ---- 1. convert weights to bf16 ----
  const int pbs[4] = {21, 37, 53, 69};   // local, global0..2
  for (int l = 0; l < 4; ++l) {
    int pb = pbs[l];
    cvt(d_in[pb + 0],  Wq[l], D * D);
    cvt(d_in[pb + 2],  Wk[l], D * D);
    cvt(d_in[pb + 4],  Wv[l], D * D);
    cvt(d_in[pb + 6],  Wo[l], D * D);
    cvt(d_in[pb + 8],  W1[l], D * NFF);
    cvt(d_in[pb + 10], W2[l], NFF * D);
  }
  cvt(d_in[3],  subjW, 2376 * 512);
  cvt(d_in[5],  objW,  2376 * 512);
  cvt(d_in[17], vrW,   12544 * 512);
  transpose_cvt_kernel<<<(256 * 1024 + 255) / 256, 256, 0, stream>>>(FP(7),  uW,  256, 1024);
  transpose_cvt_kernel<<<(256 * 1152 + 255) / 256, 256, 0, stream>>>(FP(13), c2W, 256, 1152);

  // ---- 2. subj / obj projections ----
  gather_bf16_kernel<<<1024, 256, 0, stream>>>(features, pair_idx, 0, subjA);
  gather_bf16_kernel<<<1024, 256, 0, stream>>>(features, pair_idx, 1, objA);
  gemm(subjA, subjW, FP(4), subjO, nullptr, 1024, 512, 2376, 0);
  gemm(objA,  objW,  FP(6), objO,  nullptr, 1024, 512, 2376, 0);

  // ---- 3. union 1x1 conv as GEMM ----
  union_pack_kernel<<<(1024 * 1024 * 49 + 255) / 256, 256, 0, stream>>>(union_feat, uA);
  gemm(uA, uW, FP(8), u_mat, nullptr, 50176, 256, 1024, 0);

  // ---- 4. spatial path: conv1+relu+bn+pool, im2col, conv2 GEMM ----
  conv1_pool_kernel<<<(1024 * 128 * 49 + 255) / 256, 256, 0, stream>>>(
      spatial, FP(9), FP(10), FP(11), FP(12), pooled);
  {
    long long tot = 50176LL * 1152LL;
    im2col_kernel<<<(unsigned)((tot + 255) / 256), 256, 0, stream>>>(pooled, colB);
  }
  gemm(colB, c2W, FP(14), c2out, nullptr, 50176, 256, 1152, 0);

  // ---- 5. vr = (u + bn2(relu(conv2))) @ vr_fc ----
  combine_vr_kernel<<<(1024 * 12544 + 255) / 256, 256, 0, stream>>>(
      u_mat, c2out, FP(15), FP(16), vrIn);
  gemm(vrIn, vrW, FP(18), vrO, nullptr, 1024, 512, 12544, 0);

  // ---- 6. build rel / x ----
  build_rel_kernel<<<1024, 256, 0, stream>>>(subjO, objO, vrO, FP(19), FP(20),
                                             pair_idx, pred_lab, X1, X1b);

  // ---- 7. encoder layers ----
  auto enc = [&](float* X, bf16* Xb, int li, int Bb, int S, int causal) {
    int pb = pbs[li];
    gemm(Xb, Wq[li], FP(pb + 1), Qb, nullptr, NROW, D, D, 0);
    gemm(Xb, Wk[li], FP(pb + 3), Kb, nullptr, NROW, D, D, 0);
    gemm(Xb, Wv[li], FP(pb + 5), Vb, nullptr, NROW, D, D, 0);
    attn_kernel<<<Bb * NH * S, 256, 0, stream>>>(Qb, Kb, Vb, Ob, Bb, S, causal);
    convert_f2b_kernel<<<(NROW * D + 255) / 256, 256, 0, stream>>>(Ob, Obf, NROW * D);
    gemm(Obf, Wo[li], FP(pb + 7), Pb, nullptr, NROW, D, D, 0);
    ln_res_kernel<<<NROW, 256, 0, stream>>>(X, Pb, FP(pb + 12), FP(pb + 13), Yb, Ybf, D);
    gemm(Ybf, W1[li], FP(pb + 9),  Hb, Hbf,     NROW, NFF, D, 1);
    gemm(Hbf, W2[li], FP(pb + 11), Fb, nullptr, NROW, D, NFF, 0);
    ln_res_kernel<<<NROW, 256, 0, stream>>>(Yb, Fb, FP(pb + 14), FP(pb + 15), X, Xb, D);
  };

  enc(X1, X1b, 0, NF, NP, 0);                       // local, non-causal
  pos_transpose_kernel<<<NROW, 256, 0, stream>>>(X1, X2, X2b);
  for (int g = 1; g <= 3; ++g) enc(X2, X2b, g, NP, NF, 1);  // global, causal

  // ---- 8. outputs ----
  float* out = (float*)d_out;
  float* go  = out + 26624;                         // 3072 + 6144 + 17408
  write_go_kernel<<<NROW, 256, 0, stream>>>(X2, go);
  dim3 hgrid(26, NROW);
  heads_kernel<<<hgrid, 64, 0, stream>>>(go, FP(85), FP(86), FP(87), FP(88),
                                         FP(89), FP(90), out);
}
